// MultiHeadAttention_377957122325
// MI455X (gfx1250) — matrix operand order repaired and checked by reading, not compiled, not run on hardware
//
#include <hip/hip_runtime.h>

typedef __attribute__((ext_vector_type(16))) __bf16 v16bf;
typedef __attribute__((ext_vector_type(8)))  __bf16 v8bf;
typedef __attribute__((ext_vector_type(8)))  float  v8f;

static constexpr int Bn = 2, Tn = 2048, Dn = 1024, Hn = 16, DH = 64;
static constexpr int Mn = Bn * Tn;   // 4096 rows
static constexpr int MW = Tn / 32;   // 64 mask words per row

__device__ __forceinline__ v16bf cat8(v8bf lo, v8bf hi) {
  return __builtin_shufflevector(lo, hi, 0, 1, 2, 3, 4, 5, 6, 7,
                                 8, 9, 10, 11, 12, 13, 14, 15);
}
__device__ __forceinline__ v16bf load16(const __bf16* p) {
  return cat8(*(const v8bf*)p, *(const v8bf*)(p + 8));
}
__device__ __forceinline__ v8f wmma_bf16(v16bf a, v16bf b, v8f c) {
  return __builtin_amdgcn_wmma_f32_16x16x32_bf16(false, a, false, b,
                                                 (short)0, c, false, false);
}
// A-fragment: one contiguous 16-elem run.
// lanes 0-15: K = 0..15; lanes 16-31: K = 16..31.
__device__ __forceinline__ v16bf loadA(const __bf16* row, int half) {
  return cat8(*(const v8bf*)(row + 16 * half),
              *(const v8bf*)(row + 16 * half + 8));
}

// ---- CDNA5 async global->LDS copy (ASYNCcnt path, ISA 10.7 / 15.18.3) ----
// Copies 64 bytes: 4 x GLOBAL_LOAD_ASYNC_TO_LDS_B128. INST_OFFSET applies to
// both the LDS and global addresses, so one base pair covers the chunk.
__device__ __forceinline__ void async_copy64(unsigned lds, const __bf16* g) {
  const unsigned long long ga = (unsigned long long)(size_t)g;
  asm volatile("global_load_async_to_lds_b128 %0, %1, off"
               :: "v"(lds), "v"(ga));
  asm volatile("global_load_async_to_lds_b128 %0, %1, off offset:16"
               :: "v"(lds), "v"(ga));
  asm volatile("global_load_async_to_lds_b128 %0, %1, off offset:32"
               :: "v"(lds), "v"(ga));
  asm volatile("global_load_async_to_lds_b128 %0, %1, off offset:48"
               :: "v"(lds), "v"(ga));
}
__device__ __forceinline__ void wait_async0() {
  asm volatile("s_wait_asynccnt 0x0" ::: "memory");
}
// low 32 bits of a generic pointer into LDS == LDS byte offset (flat aperture)
__device__ __forceinline__ unsigned lds_addr(const void* p) {
  return (unsigned)(size_t)p;
}

// ---------------------------------------------------------------------------
// X f32 -> bf16 row-major (8 elements / thread)
// ---------------------------------------------------------------------------
__global__ void f32_to_bf16(const float* __restrict__ in,
                            __bf16* __restrict__ out, int n) {
  const int i = (blockIdx.x * blockDim.x + threadIdx.x) * 8;
  if (i >= n) return;
  float4 x0 = *(const float4*)(in + i);
  float4 x1 = *(const float4*)(in + i + 4);
  v8bf o;
  o[0] = (__bf16)x0.x; o[1] = (__bf16)x0.y; o[2] = (__bf16)x0.z; o[3] = (__bf16)x0.w;
  o[4] = (__bf16)x1.x; o[5] = (__bf16)x1.y; o[6] = (__bf16)x1.z; o[7] = (__bf16)x1.w;
  *(v8bf*)(out + i) = o;
}

// ---------------------------------------------------------------------------
// mask int32 [B,T,T] -> packed bits [B,T,T/32] (bit k = mask element != 0)
// ---------------------------------------------------------------------------
__global__ void pack_mask(const int* __restrict__ m,
                          unsigned* __restrict__ out, int nwords) {
  const int w = blockIdx.x * blockDim.x + threadIdx.x;
  if (w >= nwords) return;
  const int* p = m + (size_t)w * 32;
  unsigned bits = 0;
#pragma unroll
  for (int i = 0; i < 8; ++i) {
    int4 v = *(const int4*)(p + i * 4);
    bits |= (v.x != 0 ? 1u : 0u) << (i * 4 + 0);
    bits |= (v.y != 0 ? 1u : 0u) << (i * 4 + 1);
    bits |= (v.z != 0 ? 1u : 0u) << (i * 4 + 2);
    bits |= (v.w != 0 ? 1u : 0u) << (i * 4 + 3);
  }
  out[w] = bits;
}

// ---------------------------------------------------------------------------
// W[K][N] f32 -> Wt[N][K] bf16 (tiled LDS transpose, 32x32 tiles)
// ---------------------------------------------------------------------------
__global__ void transpose_w(const float* __restrict__ W,
                            __bf16* __restrict__ Wt) {
  __shared__ __bf16 tile[32][33];
  const int n0 = blockIdx.x * 32, k0 = blockIdx.y * 32;
  const int tx = threadIdx.x, ty = threadIdx.y;  // 32 x 8
#pragma unroll
  for (int i = 0; i < 32; i += 8)
    tile[ty + i][tx] = (__bf16)W[(size_t)(k0 + ty + i) * Dn + n0 + tx];
  __syncthreads();
#pragma unroll
  for (int i = 0; i < 32; i += 8)
    Wt[(size_t)(n0 + ty + i) * Dn + k0 + tx] = tile[tx][ty + i];
}

// ---------------------------------------------------------------------------
// GEMM: C[M,N] = (A[M,K] @ Bt[N,K]^T + bias[N]) * alpha, bf16 WMMA, f32 accum.
// One wave per 64x32 output tile: 4 A-frags x 2 B-frags -> 8 WMMAs / K-step.
// OUT_MODE: 0 = f32 row-major, 1 = bf16 row-major, 2 = bf16 V-transposed
//           layout [B, H, dh, T].
// ---------------------------------------------------------------------------
template <int OUT_MODE>
__global__ void gemm_wmma64x32(const __bf16* __restrict__ A,
                               const __bf16* __restrict__ Bt,
                               const float* __restrict__ bias,
                               void* __restrict__ Cout,
                               int M, int N, int K, float alpha) {
  const int lane = threadIdx.x & 31;
  const int half = lane >> 4;
  const int r    = lane & 15;
  const int wave = blockIdx.x * (blockDim.x >> 5) + (threadIdx.x >> 5);
  const int ntiles = N / 32;
  const int tm = wave / ntiles, tn = wave % ntiles;
  if (tm * 64 >= M) return;
  const int mbase = tm * 64, nbase = tn * 32;

  v8f acc[4][2] = {};
  for (int kk = 0; kk < K; kk += 32) {
    v16bf af[4];
#pragma unroll
    for (int i = 0; i < 4; ++i) {
      const __bf16* arow = A + (size_t)(mbase + i * 16 + r) * K + kk;
      af[i] = loadA(arow, half);
      __builtin_prefetch(arow + 64, 0, 1);
    }
    v16bf bf[2];
#pragma unroll
    for (int jn = 0; jn < 2; ++jn) {
      const __bf16* brow = Bt + (size_t)(nbase + jn * 16 + r) * K + kk + 16 * half;
      bf[jn] = load16(brow);
      __builtin_prefetch(brow + 64, 0, 1);
    }
#pragma unroll
    for (int i = 0; i < 4; ++i)
#pragma unroll
      for (int jn = 0; jn < 2; ++jn)
        acc[i][jn] = wmma_bf16(af[i], bf[jn], acc[i][jn]);
  }

#pragma unroll
  for (int i = 0; i < 4; ++i)
#pragma unroll
    for (int jn = 0; jn < 2; ++jn) {
      const int colg = nbase + jn * 16 + r;
      const float bb = bias[colg];
#pragma unroll
      for (int j = 0; j < 8; ++j) {
        const int row = mbase + i * 16 + j + 8 * half;
        const float v = (acc[i][jn][j] + bb) * alpha;
        if (OUT_MODE == 0) {
          ((float*)Cout)[(size_t)row * N + colg] = v;
        } else if (OUT_MODE == 1) {
          ((__bf16*)Cout)[(size_t)row * N + colg] = (__bf16)v;
        } else {
          const int bb2 = row >> 11, t = row & (Tn - 1);
          const int h = colg >> 6, d = colg & (DH - 1);
          ((__bf16*)Cout)[((size_t)((bb2 * Hn + h) * DH + d)) * Tn + t] = (__bf16)v;
        }
      }
    }
}

// ---------------------------------------------------------------------------
// Fused masked attention, flash-style. Block = 4 waves = 4 q-tiles of the
// SAME (b,h). K/V 64-key tiles double-buffered in LDS via CDNA5 async
// global->LDS copies (issued before the WMMA section, waited at the barrier).
// Mask consumed as packed bits (8 x b64 per 64-key block).
// ---------------------------------------------------------------------------
__global__ void __launch_bounds__(128)
flash_attn(const __bf16* __restrict__ Q,
           const __bf16* __restrict__ Kb,
           const __bf16* __restrict__ Vt,
           const unsigned* __restrict__ Mb,
           __bf16* __restrict__ Out) {
  // padded row stride 72 (=144B) spreads LDS banks for b128 fragment reads
  __shared__ __align__(16) __bf16 kls[2][64 * 72];  // [key][dh]
  __shared__ __align__(16) __bf16 vls[2][64 * 72];  // [dh][key]
  __shared__ __align__(16) __bf16 pbuf[4][16 * 72]; // per-wave P tile
  const int tid  = threadIdx.x;
  const int lane = tid & 31, half = lane >> 4, r = lane & 15;
  const int widx = tid >> 5;
  __bf16* pw = pbuf[widx];

  int wg = blockIdx.x * 4 + widx;
  const int qt = wg % (Tn / 16); wg /= (Tn / 16);
  const int h  = wg % Hn;        wg /= Hn;
  const int b  = wg;
  const int qbase = qt * 16;

  // staging role: each thread owns half of one tile row (64B)
  const int srow = tid >> 1, spart = (tid & 1) * 32;
  const __bf16* ksrc = Kb + (size_t)(b * Tn + srow) * Dn + h * DH + spart;
  const __bf16* vsrc = Vt + (size_t)((b * Hn + h) * DH + srow) * Tn + spart;
  const unsigned kd[2] = { lds_addr(&kls[0][srow * 72 + spart]),
                           lds_addr(&kls[1][srow * 72 + spart]) };
  const unsigned vd[2] = { lds_addr(&vls[0][srow * 72 + spart]),
                           lds_addr(&vls[1][srow * 72 + spart]) };

  // prologue: stage block 0 into buffer 0 (async)
  async_copy64(kd[0], ksrc);
  async_copy64(vd[0], vsrc);

  // Q A-fragments
  const __bf16* qrow = Q + (size_t)(b * Tn + qbase + r) * Dn + h * DH;
  const v16bf qa0 = loadA(qrow, half);
  const v16bf qa1 = loadA(qrow + 32, half);

  float rmax[8], rsum[8];
#pragma unroll
  for (int j = 0; j < 8; ++j) { rmax[j] = -3.0e38f; rsum[j] = 0.0f; }
  v8f o[4] = {};

  const unsigned* mbase0 = Mb + (size_t)(b * Tn + qbase + half * 8) * MW;

  wait_async0();
  __syncthreads();

  for (int kb = 0; kb < Tn; kb += 64) {
    const int cur = (kb >> 6) & 1;
    const __bf16* kcur = kls[cur];
    const __bf16* vcur = vls[cur];
    const bool more = (kb + 64) < Tn;

    // kick off next-block async staging (hidden behind the WMMA section)
    if (more) {
      async_copy64(kd[cur ^ 1], ksrc + (size_t)(kb + 64) * Dn);
      async_copy64(vd[cur ^ 1], vsrc + kb + 64);
    }

    // packed mask words: 64 key-bits per query row
    uint2 mw[8];
#pragma unroll
    for (int j = 0; j < 8; ++j)
      mw[j] = *(const uint2*)(mbase0 + (size_t)j * MW + (kb >> 5));

    // ---- S = Q@K^T for 4 x 16 keys (from LDS) ----
    v8f st[4];
#pragma unroll
    for (int sub = 0; sub < 4; ++sub) {
      const __bf16* krow = kcur + (sub * 16 + r) * 72 + half * 16;
      v16bf k0 = load16(krow);
      v16bf k1 = load16(krow + 32);
      v8f s = {};
      s = wmma_bf16(qa0, k0, s);
      s = wmma_bf16(qa1, k1, s);
      st[sub] = s;
    }
    // mask apply: bit (sub&1)*16 + r of word (sub>>1)
#pragma unroll
    for (int j = 0; j < 8; ++j) {
      const unsigned ux = mw[j].x >> r, uy = mw[j].y >> r;
      st[0][j] = (ux & 1u)         ? st[0][j] : -3.4028235e38f;
      st[1][j] = ((ux >> 16) & 1u) ? st[1][j] : -3.4028235e38f;
      st[2][j] = (uy & 1u)         ? st[2][j] : -3.4028235e38f;
      st[3][j] = ((uy >> 16) & 1u) ? st[3][j] : -3.4028235e38f;
    }

    // ---- joint online softmax over 64 keys ----
#pragma unroll
    for (int j = 0; j < 8; ++j) {
      float v = fmaxf(fmaxf(st[0][j], st[1][j]), fmaxf(st[2][j], st[3][j]));
#pragma unroll
      for (int d = 1; d < 16; d <<= 1) v = fmaxf(v, __shfl_xor(v, d, 32));
      const float mn = fmaxf(rmax[j], v);
      const float p0 = __expf(st[0][j] - mn);
      const float p1 = __expf(st[1][j] - mn);
      const float p2 = __expf(st[2][j] - mn);
      const float p3 = __expf(st[3][j] - mn);
      const float sc = __expf(rmax[j] - mn);
      float ls = (p0 + p1) + (p2 + p3);
#pragma unroll
      for (int d = 1; d < 16; d <<= 1) ls += __shfl_xor(ls, d, 32);
      rsum[j] = rsum[j] * sc + ls;
      rmax[j] = mn;
      o[0][j] *= sc; o[1][j] *= sc; o[2][j] *= sc; o[3][j] *= sc;
      __bf16* prow = pw + (j + half * 8) * 72 + r;
      prow[0]  = (__bf16)p0;
      prow[16] = (__bf16)p1;
      prow[32] = (__bf16)p2;
      prow[48] = (__bf16)p3;
    }

    // ---- P A-fragments (LDS b128 reloads) ----
    const v16bf pa0 = loadA(pw + r * 72, half);
    const v16bf pa1 = loadA(pw + r * 72 + 32, half);

    // ---- O += P @ V (from LDS) ----
#pragma unroll
    for (int c = 0; c < 4; ++c) {
      const __bf16* vrow = vcur + (c * 16 + r) * 72 + 16 * half;
      o[c] = wmma_bf16(pa0, load16(vrow), o[c]);
      o[c] = wmma_bf16(pa1, load16(vrow + 32), o[c]);
    }

    // async staging must land before anyone reads the other buffer
    wait_async0();
    __syncthreads();
  }

  // ---- normalize, write combined heads (bf16) ----
#pragma unroll
  for (int j = 0; j < 8; ++j) {
    const float inv = 1.0f / rsum[j];
    __bf16* orow = Out + (size_t)(b * Tn + qbase + j + half * 8) * Dn + h * DH + r;
#pragma unroll
    for (int c = 0; c < 4; ++c) orow[c * 16] = (__bf16)(o[c][j] * inv);
  }
}

// ---------------------------------------------------------------------------
extern "C" void kernel_launch(void* const* d_in, const int* in_sizes, int n_in,
                              void* d_out, int out_size, void* d_ws, size_t ws_size,
                              hipStream_t stream) {
  const float* X    = (const float*)d_in[0];
  const int*   mask = (const int*)d_in[1];
  const float* Wq   = (const float*)d_in[2];
  const float* bq   = (const float*)d_in[3];
  const float* Wk   = (const float*)d_in[4];
  const float* bk   = (const float*)d_in[5];
  const float* Wv   = (const float*)d_in[6];
  const float* bv   = (const float*)d_in[7];
  const float* Wo   = (const float*)d_in[8];
  const float* bo   = (const float*)d_in[9];

  char* ws = (char*)d_ws;
  __bf16*   Xb  = (__bf16*)(ws);                        //  8 MB [M,K] bf16
  __bf16*   Wqt = (__bf16*)(ws + ((size_t) 8 << 20));   //  2 MB [N,K] bf16
  __bf16*   Wkt = (__bf16*)(ws + ((size_t)10 << 20));   //  2 MB
  __bf16*   Wvt = (__bf16*)(ws + ((size_t)12 << 20));   //  2 MB
  __bf16*   Wot = (__bf16*)(ws + ((size_t)14 << 20));   //  2 MB
  __bf16*   Qb  = (__bf16*)(ws + ((size_t)16 << 20));   //  8 MB [B*T,D]
  __bf16*   Kbf = (__bf16*)(ws + ((size_t)24 << 20));   //  8 MB [B*T,D]
  __bf16*   Vtb = (__bf16*)(ws + ((size_t)32 << 20));   //  8 MB [B,H,dh,T]
  __bf16*   Cmb = (__bf16*)(ws + ((size_t)40 << 20));   //  8 MB [B*T,D]
  unsigned* Mb  = (unsigned*)(ws + ((size_t)48 << 20)); //  1 MB packed mask

  // Stage 0: operand conditioning
  const int nX = Mn * Dn;
  hipLaunchKernelGGL(f32_to_bf16, dim3(nX / (256 * 8)), dim3(256), 0, stream,
                     X, Xb, nX);
  const int nMW = Bn * Tn * MW;  // 262144 packed words
  hipLaunchKernelGGL(pack_mask, dim3(nMW / 256), dim3(256), 0, stream,
                     mask, Mb, nMW);
  dim3 tgrid(Dn / 32, Dn / 32), tblk(32, 8);
  hipLaunchKernelGGL(transpose_w, tgrid, tblk, 0, stream, Wq, Wqt);
  hipLaunchKernelGGL(transpose_w, tgrid, tblk, 0, stream, Wk, Wkt);
  hipLaunchKernelGGL(transpose_w, tgrid, tblk, 0, stream, Wv, Wvt);
  hipLaunchKernelGGL(transpose_w, tgrid, tblk, 0, stream, Wo, Wot);

  // Stage 1: projections (Q pre-scaled by dh^-0.5; V stored transposed)
  const int gtiles = (Mn / 64) * (Dn / 32);
  dim3 ggrd(gtiles / 8), gblk(256);
  hipLaunchKernelGGL((gemm_wmma64x32<1>), ggrd, gblk, 0, stream,
                     Xb, Wqt, bq, (void*)Qb, Mn, Dn, Dn, 0.125f);
  hipLaunchKernelGGL((gemm_wmma64x32<1>), ggrd, gblk, 0, stream,
                     Xb, Wkt, bk, (void*)Kbf, Mn, Dn, Dn, 1.0f);
  hipLaunchKernelGGL((gemm_wmma64x32<2>), ggrd, gblk, 0, stream,
                     Xb, Wvt, bv, (void*)Vtb, Mn, Dn, Dn, 1.0f);

  // Stage 2: fused masked attention -> combined [B,T,D] bf16
  const int awaves = Bn * Hn * (Tn / 16);
  hipLaunchKernelGGL(flash_attn, dim3(awaves / 4), dim3(128), 0, stream,
                     Qb, Kbf, Vtb, Mb, Cmb);

  // Stage 3: output projection -> f32 d_out
  hipLaunchKernelGGL((gemm_wmma64x32<0>), ggrd, gblk, 0, stream,
                     Cmb, Wot, bo, d_out, Mn, Dn, Dn, 1.0f);

  (void)in_sizes; (void)n_in; (void)out_size; (void)ws_size;
}
